// Transformer_46995532153103
// MI455X (gfx1250) — matrix operand order repaired and checked by reading, not compiled, not run on hardware
//
#include <hip/hip_runtime.h>

typedef __attribute__((ext_vector_type(16))) _Float16 v16h;
typedef __attribute__((ext_vector_type(8)))  float    v8f;
typedef __attribute__((ext_vector_type(4)))  int      v4i;

#define AS1 __attribute__((address_space(1)))
#define AS3 __attribute__((address_space(3)))

// Model dimensions (fixed by the reference)
constexpr int kV  = 32000;
constexpr int kE  = 512;
constexpr int kH  = 8;
constexpr int kD  = 64;     // kE / kH
constexpr int kT  = 1024;   // both src and tgt sequence length
constexpr int kB  = 2;
constexpr int kM  = kB * kT;   // 2048 activation rows
constexpr int kLE = 4;
constexpr int kLD = 4;

// ---------------------------------------------------------------------------
// Input index map: {src, tgt, params-pytree(alphabetical flatten)}
// ---------------------------------------------------------------------------
enum {
  IN_SRC = 0, IN_TGT = 1,
  P_DEC_CA_KB, P_DEC_CA_KW, P_DEC_CA_PB, P_DEC_CA_PW,
  P_DEC_CA_QB, P_DEC_CA_QW, P_DEC_CA_VB, P_DEC_CA_VW,
  P_DEC_LN1B, P_DEC_LN1G, P_DEC_LN2B, P_DEC_LN2G, P_DEC_LN3B, P_DEC_LN3G,
  P_DEC_FCB, P_DEC_FCW, P_DEC_PJB, P_DEC_PJW,
  P_DEC_SA_KB, P_DEC_SA_KW, P_DEC_SA_PB, P_DEC_SA_PW,
  P_DEC_SA_QB, P_DEC_SA_QW, P_DEC_SA_VB, P_DEC_SA_VW,
  P_DEC_LNFB, P_DEC_LNFG, P_DEC_WPE, P_DEC_WTE,
  P_ENC_KB, P_ENC_KW, P_ENC_PB, P_ENC_PW,
  P_ENC_QB, P_ENC_QW, P_ENC_VB, P_ENC_VW,
  P_ENC_LN1B, P_ENC_LN1G, P_ENC_LN2B, P_ENC_LN2G,
  P_ENC_FCB, P_ENC_FCW, P_ENC_PJB, P_ENC_PJW,
  P_ENC_LNFB, P_ENC_LNFG, P_ENC_WPE, P_ENC_WTE,
  P_LMHEAD
};

// ---------------------------------------------------------------------------
// CDNA5 async global->LDS copy (16B per lane) + ASYNCcnt waits
// ---------------------------------------------------------------------------
__device__ inline void async_copy16(const float* g, float* l) {
#if __has_builtin(__builtin_amdgcn_global_load_async_to_lds_b128)
  __builtin_amdgcn_global_load_async_to_lds_b128((AS1 v4i*)g, (AS3 v4i*)l, 0, 0);
#else
  unsigned l32 = (unsigned)(unsigned long long)(AS3 void*)l;
  asm volatile("global_load_async_to_lds_b128 %0, %1, off"
               :: "v"(l32), "v"(g) : "memory");
#endif
}

__device__ inline void wait_async_le6() {
#if __has_builtin(__builtin_amdgcn_s_wait_asynccnt)
  __builtin_amdgcn_s_wait_asynccnt(6);
#else
  asm volatile("s_wait_asynccnt 0x6" ::: "memory");
#endif
}

__device__ inline void wait_async_le0() {
#if __has_builtin(__builtin_amdgcn_s_wait_asynccnt)
  __builtin_amdgcn_s_wait_asynccnt(0);
#else
  asm volatile("s_wait_asynccnt 0x0" ::: "memory");
#endif
}

// ---------------------------------------------------------------------------
// Embedding: x[b,t,:] = wte[tok[b,t],:] + wpe[t,:]
// ---------------------------------------------------------------------------
__global__ void embed_kernel(const int* __restrict__ tok,
                             const float* __restrict__ wte,
                             const float* __restrict__ wpe,
                             float* __restrict__ X, int T) {
  int idx = blockIdx.x * blockDim.x + threadIdx.x;  // over B*T*E
  int c  = idx & (kE - 1);
  int bt = idx / kE;
  int t  = bt % T;
  int tk = tok[bt];
  X[idx] = wte[(size_t)tk * kE + c] + wpe[(size_t)t * kE + c];
}

// ---------------------------------------------------------------------------
// LayerNorm over E=512, one wave32 per row, shfl reductions
// ---------------------------------------------------------------------------
__global__ __launch_bounds__(32)
void layernorm_kernel(const float* __restrict__ X,
                      const float* __restrict__ g,
                      const float* __restrict__ b,
                      float* __restrict__ Y) {
  const int row  = blockIdx.x;
  const int lane = threadIdx.x & 31;
  const float* x = X + (size_t)row * kE;
  float vals[16];
  float s = 0.f;
#pragma unroll
  for (int i = 0; i < 16; ++i) { vals[i] = x[lane + 32 * i]; s += vals[i]; }
#pragma unroll
  for (int m = 16; m >= 1; m >>= 1) s += __shfl_xor(s, m, 32);
  const float mu = s * (1.0f / kE);
  float vs = 0.f;
#pragma unroll
  for (int i = 0; i < 16; ++i) { float d = vals[i] - mu; vs += d * d; }
#pragma unroll
  for (int m = 16; m >= 1; m >>= 1) vs += __shfl_xor(vs, m, 32);
  const float inv = rsqrtf(vs * (1.0f / kE) + 1e-5f);
  float* y = Y + (size_t)row * kE;
#pragma unroll
  for (int i = 0; i < 16; ++i) {
    int c = lane + 32 * i;
    y[c] = (vals[i] - mu) * inv * g[c] + b[c];
  }
}

// ---------------------------------------------------------------------------
// Tiled WMMA GEMM with async-LDS double buffering.
//   C[M,N] = epilogue(A[M,K] @ W[K,N] + bias [, +residual])
// Workgroup = 256 threads (8 waves) -> 128x64 block tile.
// Wave (wm,wn) computes a 32x32 register block = 2x2 WMMA tiles.
// K staged in 32-wide slices: A slice 128x32 fp32 (16KB, 4 async b128/lane),
// B slice 32x64 fp32 (8KB, 2 async b128/lane); double buffered (48KB LDS),
// next slice DMA overlapped with current slice WMMAs via s_wait_asynccnt.
// ---------------------------------------------------------------------------
template <int ACT, bool RES>
__global__ __launch_bounds__(256)
void gemm_wmma_kernel(const float* __restrict__ A, const float* __restrict__ W,
                      const float* __restrict__ bias,
                      const float* __restrict__ Rsd,
                      float* __restrict__ C, int M, int N, int K) {
  __shared__ __align__(16) float Abuf[2][128 * 32];
  __shared__ __align__(16) float Bbuf[2][32 * 64];

  const int tid    = threadIdx.x;
  const int lane   = tid & 31;
  const int wave   = tid >> 5;
  const int wm     = wave & 3;        // M quadrant (rows wm*32..wm*32+31)
  const int wn     = wave >> 2;       // N half    (cols wn*32..wn*32+31)
  const int half16 = lane & 15;
  const int rbase  = (lane >> 4) * 8; // C rows this lane owns
  const int akb    = (lane >> 4) * 16; // A K-split: k = 16*(lane/16) + s (contiguous)
  const int bkb    = (lane >> 4) * 16;// B K-split: k = 16*(lane/16) + s, same as A
  const int nb = blockIdx.x * 64;
  const int mb = blockIdx.y * 128;

  auto stage = [&](int buf, int kk) {
#pragma unroll
    for (int it = 0; it < 4; ++it) {          // A: 128x32 fp32
      int c = it * 256 + tid;
      int row = c >> 3, co = (c & 7) * 4;
      async_copy16(A + (size_t)(mb + row) * K + kk + co, &Abuf[buf][row * 32 + co]);
    }
#pragma unroll
    for (int it = 0; it < 2; ++it) {          // B: 32x64 fp32
      int c = it * 256 + tid;
      int row = c >> 4, co = (c & 15) * 4;
      async_copy16(W + (size_t)(kk + row) * N + nb + co, &Bbuf[buf][row * 64 + co]);
    }
  };

  v8f acc[2][2] = {};
  const int KT = K / 32;
  stage(0, 0);
  for (int kt = 0; kt < KT; ++kt) {
    const int buf = kt & 1;
    if (kt + 1 < KT) { stage(buf ^ 1, (kt + 1) * 32); wait_async_le6(); }
    else             { wait_async_le0(); }
    __syncthreads();

    const float* Ab = Abuf[buf];
    const float* Bb = Bbuf[buf];
    v16h afr[2], bfr[2];
#pragma unroll
    for (int mt = 0; mt < 2; ++mt) {
      const float* ar = Ab + (wm * 32 + mt * 16 + half16) * 32;
#pragma unroll
      for (int i = 0; i < 8; ++i) {
        afr[mt][i]     = (_Float16)ar[akb + i];
        afr[mt][8 + i] = (_Float16)ar[8 + akb + i];
      }
    }
#pragma unroll
    for (int nt = 0; nt < 2; ++nt) {
      const float* br = Bb + wn * 32 + nt * 16 + half16;
#pragma unroll
      for (int i = 0; i < 16; ++i) bfr[nt][i] = (_Float16)br[(bkb + i) * 64];
    }
#pragma unroll
    for (int mt = 0; mt < 2; ++mt)
#pragma unroll
      for (int nt = 0; nt < 2; ++nt)
        acc[mt][nt] = __builtin_amdgcn_wmma_f32_16x16x32_f16(
            false, afr[mt], false, bfr[nt], (short)0, acc[mt][nt], false, false);
    __syncthreads();
  }

#pragma unroll
  for (int mt = 0; mt < 2; ++mt)
#pragma unroll
    for (int nt = 0; nt < 2; ++nt) {
      const int col = nb + wn * 32 + nt * 16 + half16;
      const float bval = bias ? bias[col] : 0.0f;
#pragma unroll
      for (int r = 0; r < 8; ++r) {
        const int row = mb + wm * 32 + mt * 16 + rbase + r;
        float vv = acc[mt][nt][r] + bval;
        if (ACT == 1) vv = 0.5f * vv * (1.0f + erff(vv * 0.70710678118654752f));
        if (RES) vv += Rsd[(size_t)row * N + col];
        C[(size_t)row * N + col] = vv;
      }
    }
}

// ---------------------------------------------------------------------------
// Fused flash-style attention.  One wave32 per (batch, head, 16-query tile).
// QK^T via WMMA (two K=32 steps over D=64), online softmax in fp32, P routed
// through LDS (C-layout -> A-layout), P.V via WMMA.
// ---------------------------------------------------------------------------
__global__ __launch_bounds__(32)
void attn_kernel(const float* __restrict__ Q, const float* __restrict__ K,
                 const float* __restrict__ V, float* __restrict__ O,
                 int Tq, int Tk, int causal) {
  __shared__ _Float16 Pt[16 * 32];

  const int lane   = threadIdx.x & 31;
  const int half16 = lane & 15;
  const int rbase  = (lane >> 4) * 8;
  const int akb    = (lane >> 4) * 16;
  const int bkb    = (lane >> 4) * 16;

  const int qt   = blockIdx.x;
  const int h    = blockIdx.y;
  const int bz   = blockIdx.z;
  const int col0 = h * kD;
  const int qrow0 = bz * Tq + qt * 16;
  const int krow0 = bz * Tk;

  v16h aq0, aq1;
  {
    const float* qrow = Q + (size_t)(qrow0 + half16) * kE + col0;
    const float sc = 0.125f;  // 1/sqrt(64)
#pragma unroll
    for (int i = 0; i < 8; ++i) {
      aq0[i]     = (_Float16)(qrow[akb + i] * sc);
      aq0[8 + i] = (_Float16)(qrow[8 + akb + i] * sc);
      aq1[i]     = (_Float16)(qrow[32 + akb + i] * sc);
      aq1[8 + i] = (_Float16)(qrow[40 + akb + i] * sc);
    }
  }

  float mrun[8], lrun[8];
  v8f acc0 = {}, acc1 = {}, acc2 = {}, acc3 = {};
#pragma unroll
  for (int r = 0; r < 8; ++r) { mrun[r] = -1e30f; lrun[r] = 0.f; }

  const int kend = causal ? (qt * 16 + 16) : Tk;
  for (int kk = 0; kk < kend; kk += 32) {
    v8f s0 = {}, s1 = {};
    {
      v16h b0, b1;
      const float* krow = K + (size_t)(krow0 + kk + half16) * kE + col0;
#pragma unroll
      for (int i = 0; i < 16; ++i) {
        b0[i] = (_Float16)krow[bkb + i];
        b1[i] = (_Float16)krow[32 + bkb + i];
      }
      s0 = __builtin_amdgcn_wmma_f32_16x16x32_f16(false, aq0, false, b0,
                                                  (short)0, s0, false, false);
      s0 = __builtin_amdgcn_wmma_f32_16x16x32_f16(false, aq1, false, b1,
                                                  (short)0, s0, false, false);
    }
    {
      v16h b0, b1;
      const float* krow = K + (size_t)(krow0 + kk + 16 + half16) * kE + col0;
#pragma unroll
      for (int i = 0; i < 16; ++i) {
        b0[i] = (_Float16)krow[bkb + i];
        b1[i] = (_Float16)krow[32 + bkb + i];
      }
      s1 = __builtin_amdgcn_wmma_f32_16x16x32_f16(false, aq0, false, b0,
                                                  (short)0, s1, false, false);
      s1 = __builtin_amdgcn_wmma_f32_16x16x32_f16(false, aq1, false, b1,
                                                  (short)0, s1, false, false);
    }

    if (causal) {
      const int qabs = qt * 16 + rbase;
#pragma unroll
      for (int r = 0; r < 8; ++r) {
        if (kk + half16 > qabs + r)      s0[r] = -1e30f;
        if (kk + 16 + half16 > qabs + r) s1[r] = -1e30f;
      }
    }

    float alpha[8], p0[8], p1[8];
#pragma unroll
    for (int r = 0; r < 8; ++r) {
      float mx = fmaxf(s0[r], s1[r]);
#pragma unroll
      for (int mk = 8; mk >= 1; mk >>= 1) mx = fmaxf(mx, __shfl_xor(mx, mk, 32));
      const float mnew = fmaxf(mrun[r], mx);
      alpha[r] = __expf(mrun[r] - mnew);
      mrun[r]  = mnew;
      p0[r] = __expf(s0[r] - mnew);
      p1[r] = __expf(s1[r] - mnew);
      float rs = p0[r] + p1[r];
#pragma unroll
      for (int mk = 8; mk >= 1; mk >>= 1) rs += __shfl_xor(rs, mk, 32);
      lrun[r] = lrun[r] * alpha[r] + rs;
      acc0[r] *= alpha[r]; acc1[r] *= alpha[r];
      acc2[r] *= alpha[r]; acc3[r] *= alpha[r];
    }

    __syncthreads();
#pragma unroll
    for (int r = 0; r < 8; ++r) {
      Pt[(rbase + r) * 32 + half16]      = (_Float16)p0[r];
      Pt[(rbase + r) * 32 + 16 + half16] = (_Float16)p1[r];
    }
    __syncthreads();
    v16h pa;
#pragma unroll
    for (int i = 0; i < 8; ++i) {
      pa[i]     = Pt[half16 * 32 + akb + i];
      pa[8 + i] = Pt[half16 * 32 + 8 + akb + i];
    }

    const size_t vrow0 = (size_t)(krow0 + kk + ((lane >> 4) * 16));
    v16h bv;
#pragma unroll
    for (int i = 0; i < 16; ++i)
      bv[i] = (_Float16)V[(vrow0 + i) * kE + col0 + half16];
    acc0 = __builtin_amdgcn_wmma_f32_16x16x32_f16(false, pa, false, bv,
                                                  (short)0, acc0, false, false);
#pragma unroll
    for (int i = 0; i < 16; ++i)
      bv[i] = (_Float16)V[(vrow0 + i) * kE + col0 + 16 + half16];
    acc1 = __builtin_amdgcn_wmma_f32_16x16x32_f16(false, pa, false, bv,
                                                  (short)0, acc1, false, false);
#pragma unroll
    for (int i = 0; i < 16; ++i)
      bv[i] = (_Float16)V[(vrow0 + i) * kE + col0 + 32 + half16];
    acc2 = __builtin_amdgcn_wmma_f32_16x16x32_f16(false, pa, false, bv,
                                                  (short)0, acc2, false, false);
#pragma unroll
    for (int i = 0; i < 16; ++i)
      bv[i] = (_Float16)V[(vrow0 + i) * kE + col0 + 48 + half16];
    acc3 = __builtin_amdgcn_wmma_f32_16x16x32_f16(false, pa, false, bv,
                                                  (short)0, acc3, false, false);
  }

#pragma unroll
  for (int r = 0; r < 8; ++r) {
    const float invl = 1.0f / lrun[r];
    float* orow = O + (size_t)(qrow0 + rbase + r) * kE + col0;
    orow[half16]      = acc0[r] * invl;
    orow[16 + half16] = acc1[r] * invl;
    orow[32 + half16] = acc2[r] * invl;
    orow[48 + half16] = acc3[r] * invl;
  }
}

// ---------------------------------------------------------------------------
// Host-side orchestration
// ---------------------------------------------------------------------------
static inline void launch_gemm(hipStream_t s, const float* A, const float* W,
                               const float* bias, const float* rsd, float* C,
                               int M, int N, int K, int gelu) {
  dim3 g(N / 64, M / 128), b(256);
  if (gelu) {
    if (rsd) hipLaunchKernelGGL((gemm_wmma_kernel<1, true>),  g, b, 0, s, A, W, bias, rsd, C, M, N, K);
    else     hipLaunchKernelGGL((gemm_wmma_kernel<1, false>), g, b, 0, s, A, W, bias, rsd, C, M, N, K);
  } else {
    if (rsd) hipLaunchKernelGGL((gemm_wmma_kernel<0, true>),  g, b, 0, s, A, W, bias, rsd, C, M, N, K);
    else     hipLaunchKernelGGL((gemm_wmma_kernel<0, false>), g, b, 0, s, A, W, bias, rsd, C, M, N, K);
  }
}

extern "C" void kernel_launch(void* const* d_in, const int* in_sizes, int n_in,
                              void* d_out, int out_size, void* d_ws, size_t ws_size,
                              hipStream_t stream) {
  auto F = [&](int i) { return (const float*)d_in[i]; };
  const int* src = (const int*)d_in[IN_SRC];
  const int* tgt = (const int*)d_in[IN_TGT];

  const size_t ME = (size_t)kM * kE;
  float* x   = (float*)d_ws;       // [M,E] running activations
  float* h   = x   + ME;           // [M,E] LN output
  float* q   = h   + ME;           // [M,E]
  float* k   = q   + ME;           // [M,E]
  float* v   = k   + ME;           // [M,E]
  float* y   = v   + ME;           // [M,E] attention output
  float* enc = y   + ME;           // [M,E] encoder output
  float* mid = enc + ME;           // [M,4E] MLP hidden

  const dim3 attn_grid(kT / 16, kH, kB), wave(32);
  const int embed_blocks = (int)(ME / 256);

  // ===================== Encoder =====================
  embed_kernel<<<embed_blocks, 256, 0, stream>>>(src, F(P_ENC_WTE), F(P_ENC_WPE), x, kT);
  for (int l = 0; l < kLE; ++l) {
    const size_t oE = (size_t)l * kE, oEE = (size_t)l * kE * kE;
    const size_t o4E = (size_t)l * 4 * kE, oE4E = (size_t)l * kE * 4 * kE;
    layernorm_kernel<<<kM, wave, 0, stream>>>(x, F(P_ENC_LN1G) + oE, F(P_ENC_LN1B) + oE, h);
    launch_gemm(stream, h, F(P_ENC_QW) + oEE, F(P_ENC_QB) + oE, nullptr, q, kM, kE, kE, 0);
    launch_gemm(stream, h, F(P_ENC_KW) + oEE, F(P_ENC_KB) + oE, nullptr, k, kM, kE, kE, 0);
    launch_gemm(stream, h, F(P_ENC_VW) + oEE, F(P_ENC_VB) + oE, nullptr, v, kM, kE, kE, 0);
    attn_kernel<<<attn_grid, wave, 0, stream>>>(q, k, v, y, kT, kT, 0);
    launch_gemm(stream, y, F(P_ENC_PW) + oEE, F(P_ENC_PB) + oE, x, x, kM, kE, kE, 0);
    layernorm_kernel<<<kM, wave, 0, stream>>>(x, F(P_ENC_LN2G) + oE, F(P_ENC_LN2B) + oE, h);
    launch_gemm(stream, h, F(P_ENC_FCW) + oE4E, F(P_ENC_FCB) + o4E, nullptr, mid, kM, 4 * kE, kE, 1);
    launch_gemm(stream, mid, F(P_ENC_PJW) + oE4E, F(P_ENC_PJB) + oE, x, x, kM, kE, 4 * kE, 0);
  }
  layernorm_kernel<<<kM, wave, 0, stream>>>(x, F(P_ENC_LNFG), F(P_ENC_LNFB), enc);

  // ===================== Decoder =====================
  embed_kernel<<<embed_blocks, 256, 0, stream>>>(tgt, F(P_DEC_WTE), F(P_DEC_WPE), x, kT);
  for (int l = 0; l < kLD; ++l) {
    const size_t oE = (size_t)l * kE, oEE = (size_t)l * kE * kE;
    const size_t o4E = (size_t)l * 4 * kE, oE4E = (size_t)l * kE * 4 * kE;
    // causal self-attention
    layernorm_kernel<<<kM, wave, 0, stream>>>(x, F(P_DEC_LN1G) + oE, F(P_DEC_LN1B) + oE, h);
    launch_gemm(stream, h, F(P_DEC_SA_QW) + oEE, F(P_DEC_SA_QB) + oE, nullptr, q, kM, kE, kE, 0);
    launch_gemm(stream, h, F(P_DEC_SA_KW) + oEE, F(P_DEC_SA_KB) + oE, nullptr, k, kM, kE, kE, 0);
    launch_gemm(stream, h, F(P_DEC_SA_VW) + oEE, F(P_DEC_SA_VB) + oE, nullptr, v, kM, kE, kE, 0);
    attn_kernel<<<attn_grid, wave, 0, stream>>>(q, k, v, y, kT, kT, 1);
    launch_gemm(stream, y, F(P_DEC_SA_PW) + oEE, F(P_DEC_SA_PB) + oE, x, x, kM, kE, kE, 0);
    // cross-attention against encoder output
    layernorm_kernel<<<kM, wave, 0, stream>>>(x, F(P_DEC_LN2G) + oE, F(P_DEC_LN2B) + oE, h);
    launch_gemm(stream, h,   F(P_DEC_CA_QW) + oEE, F(P_DEC_CA_QB) + oE, nullptr, q, kM, kE, kE, 0);
    launch_gemm(stream, enc, F(P_DEC_CA_KW) + oEE, F(P_DEC_CA_KB) + oE, nullptr, k, kM, kE, kE, 0);
    launch_gemm(stream, enc, F(P_DEC_CA_VW) + oEE, F(P_DEC_CA_VB) + oE, nullptr, v, kM, kE, kE, 0);
    attn_kernel<<<attn_grid, wave, 0, stream>>>(q, k, v, y, kT, kT, 0);
    launch_gemm(stream, y, F(P_DEC_CA_PW) + oEE, F(P_DEC_CA_PB) + oE, x, x, kM, kE, kE, 0);
    // MLP
    layernorm_kernel<<<kM, wave, 0, stream>>>(x, F(P_DEC_LN3G) + oE, F(P_DEC_LN3B) + oE, h);
    launch_gemm(stream, h, F(P_DEC_FCW) + oE4E, F(P_DEC_FCB) + o4E, nullptr, mid, kM, 4 * kE, kE, 1);
    launch_gemm(stream, mid, F(P_DEC_PJW) + oE4E, F(P_DEC_PJB) + oE, x, x, kM, kE, 4 * kE, 0);
  }
  layernorm_kernel<<<kM, wave, 0, stream>>>(x, F(P_DEC_LNFG), F(P_DEC_LNFB), h);

  // ===================== LM head =====================
  launch_gemm(stream, h, F(P_LMHEAD), nullptr, nullptr, (float*)d_out, kM, kV, kE, 0);
}
